// StackedGridModel_47399259079382
// MI455X (gfx1250) — matrix operand order repaired and checked by reading, not compiled, not run on hardware
//
#include <hip/hip_runtime.h>

#define DEPTH_N 2
#define SRC_N   32
#define TRG_N   32
#define B_DIM   64
#define H_DIM   256

typedef __attribute__((ext_vector_type(16))) __bf16 v16bf;
typedef __attribute__((ext_vector_type(8)))  __bf16 v8bf;
typedef __attribute__((ext_vector_type(8)))  float  v8f;

// Scalar f32->bf16 RNE (prologue only; hot path uses v_cvt_pk_bf16_f32).
static __device__ __forceinline__ unsigned short f2bf(float f) {
    unsigned int u = __float_as_uint(f);
    unsigned int r = (u + 0x7FFFu + ((u >> 16) & 1u)) >> 16;
    return (unsigned short)r;
}

// Hardware packed conversion: dst[15:0]=bf16(lo), dst[31:16]=bf16(hi).
static __device__ __forceinline__ unsigned int cvt_pk_bf16(float lo, float hi) {
    unsigned int r;
    asm("v_cvt_pk_bf16_f32 %0, %1, %2" : "=v"(r) : "v"(lo), "v"(hi));
    return r;
}

// Hardware tanh (CDNA5 TRANS op). Builtin if available (compiler handles the
// TRANS hazard); else inline asm with a trailing v_nop per ISA hazard rule.
#if defined(__has_builtin) && __has_builtin(__builtin_amdgcn_tanhf)
#define HTANH(x) __builtin_amdgcn_tanhf(x)
#elif defined(__has_builtin) && __has_builtin(__builtin_amdgcn_tanh_f32)
#define HTANH(x) __builtin_amdgcn_tanh_f32(x)
#else
static __device__ __forceinline__ float htanh_asm(float x) {
    float y;
    asm("v_tanh_f32 %0, %1\n\tv_nop" : "=v"(y) : "v"(x));
    return y;
}
#define HTANH(x) htanh_asm(x)
#endif

// ---------------------------------------------------------------------------
// Prologue: W [d][K][N], U [d][2K][N] (f32) -> bf16 transposed [d][mat][N][K].
// mat 0 = W, mat 1 = U_top (U[:256]), mat 2 = U_left (U[256:]).
// ---------------------------------------------------------------------------
__global__ void prep_weights_kernel(const float* __restrict__ W,
                                    const float* __restrict__ U,
                                    unsigned short* __restrict__ wt) {
    int idx = blockIdx.x * blockDim.x + threadIdx.x;
    const int total = DEPTH_N * 3 * H_DIM * H_DIM;
    if (idx >= total) return;
    int k   = idx & (H_DIM - 1);
    int n   = (idx >> 8) & (H_DIM - 1);
    int dm  = idx >> 16;          // 0..5
    int mat = dm % 3;
    int d   = dm / 3;
    float v;
    if (mat == 0)       v = W[(size_t)d * H_DIM * H_DIM + (size_t)k * H_DIM + n];
    else if (mat == 1)  v = U[(size_t)d * 2 * H_DIM * H_DIM + (size_t)k * H_DIM + n];
    else                v = U[(size_t)d * 2 * H_DIM * H_DIM + (size_t)(H_DIM + k) * H_DIM + n];
    wt[idx] = f2bf(v);
}

// ---------------------------------------------------------------------------
// One workgroup = one grid cell (i,j) at depth d.
// 512 threads = 16 waves; wave w owns N-tile w (16 cols) x 4 M-tiles (64 rows).
// Two 32KB LDS stages hold two bf16 [B][H] activation matrices at once so each
// B-fragment (weights, L2-resident) feeds 8 WMMAs per load.
// ---------------------------------------------------------------------------
__global__ __launch_bounds__(512)
void grid_cell_kernel(const float* __restrict__ src,
                      const float* __restrict__ trg,
                      const float* __restrict__ bvec,        // [DEPTH][H]
                      const unsigned short* __restrict__ wt, // bf16 [DEPTH][3][N][K]
                      float* __restrict__ out,               // [D][S][T][2][B][H]
                      int depth, int step, int i_min) {
    __shared__ __align__(16) unsigned short smemA[B_DIM * H_DIM]; // 32 KB
    __shared__ __align__(16) unsigned short smemB[B_DIM * H_DIM]; // 32 KB

    const int i = i_min + (int)blockIdx.x;
    const int j = step - i;
    const int tid    = (int)threadIdx.x;
    const int wave   = tid >> 5;
    const int lane   = tid & 31;
    const int lane16 = lane & 15;
    const int half   = lane >> 4;
    const int n0     = wave * 16;

    const size_t MAT = (size_t)B_DIM * H_DIM;
    const size_t cell = (((size_t)depth * SRC_N + i) * TRG_N + j) * 2 * MAT;

    const float* xg = (depth == 0)
        ? (src + (size_t)i * MAT)
        : (out + ((((size_t)(depth - 1) * SRC_N + i) * TRG_N + j) * 2 + 0) * MAT);
    const float* yg = (depth == 0)
        ? (trg + (size_t)j * MAT)
        : (out + ((((size_t)(depth - 1) * SRC_N + i) * TRG_N + j) * 2 + 1) * MAT);
    const float* top  = (i > 0)
        ? (out + ((((size_t)depth * SRC_N + (i - 1)) * TRG_N + j) * 2 + 0) * MAT) : nullptr;
    const float* left = (j > 0)
        ? (out + ((((size_t)depth * SRC_N + i) * TRG_N + (j - 1)) * 2 + 0) * MAT) : nullptr;

    const unsigned short* wtW = wt + ((size_t)depth * 3 + 0) * H_DIM * H_DIM;
    const unsigned short* wtT = wt + ((size_t)depth * 3 + 1) * H_DIM * H_DIM;
    const unsigned short* wtL = wt + ((size_t)depth * 3 + 2) * H_DIM * H_DIM;

    v8f acc_x[4], acc_y[4], acc_r[4];
#pragma unroll
    for (int mt = 0; mt < 4; ++mt) {
        v8f z = {};
        acc_x[mt] = z; acc_y[mt] = z; acc_r[mt] = z;
    }

    // Cooperative f32 -> bf16 staging of one [B][H] matrix (or zeros) into LDS.
    auto stage = [&](const float* __restrict__ p, unsigned short* dst) {
        uint2* s2 = reinterpret_cast<uint2*>(dst);
        if (p) {
            const float4* p4 = reinterpret_cast<const float4*>(p);
#pragma unroll
            for (int it = 0; it < 8; ++it) {
                int idx = it * 512 + tid;            // 4096 float4 total
                float4 v = p4[idx];
                uint2 pk;
                pk.x = cvt_pk_bf16(v.x, v.y);
                pk.y = cvt_pk_bf16(v.z, v.w);
                s2[idx] = pk;
            }
        } else {
            uint2 z; z.x = 0u; z.y = 0u;
#pragma unroll
            for (int it = 0; it < 8; ++it) s2[it * 512 + tid] = z;
        }
    };

    // Load a v16bf A-fragment per ISA layout: lane -> row M = mbase+lane16,
    // K chunks at k0+8*half and k0+16+8*half.
    auto loadA = [&](const unsigned short* s, int mbase, int k0) -> v16bf {
        const unsigned short* ap = s + (size_t)(mbase + lane16) * H_DIM + k0 + 8 * half;
        v8bf a0 = *reinterpret_cast<const v8bf*>(ap);
        v8bf a1 = *reinterpret_cast<const v8bf*>(ap + 16);
        return __builtin_shufflevector(a0, a1, 0, 1, 2, 3, 4, 5, 6, 7,
                                               8, 9, 10, 11, 12, 13, 14, 15);
    };
    // Load a v16bf B-fragment: lane -> column N = n0+lane16, K chunks at k0+8*half and k0+16+8*half.
    auto loadB = [&](const unsigned short* wb, int k0) -> v16bf {
        const unsigned short* bp = wb + (size_t)(n0 + lane16) * H_DIM + k0 + 8 * half;
        v8bf b0 = *reinterpret_cast<const v8bf*>(bp);
        v8bf b1 = *reinterpret_cast<const v8bf*>(bp + 16);
        return __builtin_shufflevector(b0, b1, 0, 1, 2, 3, 4, 5, 6, 7,
                                               8, 9, 10, 11, 12, 13, 14, 15);
    };

    // ---- Phase 1: acc_x += xg@W, acc_y += yg@W (shared B-fragment) ----
    stage(xg, smemA);
    stage(yg, smemB);
    __syncthreads();
#pragma unroll 2
    for (int kc = 0; kc < 8; ++kc) {
        const int k0 = kc * 32;
        v16bf bw = loadB(wtW, k0);
#pragma unroll
        for (int mt = 0; mt < 4; ++mt) {
            v16bf ax = loadA(smemA, mt * 16, k0);
            acc_x[mt] = __builtin_amdgcn_wmma_f32_16x16x32_bf16(
                false, ax, false, bw, (short)0, acc_x[mt], false, false);
            v16bf ay = loadA(smemB, mt * 16, k0);
            acc_y[mt] = __builtin_amdgcn_wmma_f32_16x16x32_bf16(
                false, ay, false, bw, (short)0, acc_y[mt], false, false);
        }
    }
    __syncthreads();

    // ---- Phase 2: acc_r += top@U_top + left@U_left ----
    stage(top, smemA);
    stage(left, smemB);
    __syncthreads();
#pragma unroll 2
    for (int kc = 0; kc < 8; ++kc) {
        const int k0 = kc * 32;
        v16bf bt = loadB(wtT, k0);
        v16bf bl = loadB(wtL, k0);
#pragma unroll
        for (int mt = 0; mt < 4; ++mt) {
            v16bf at = loadA(smemA, mt * 16, k0);
            acc_r[mt] = __builtin_amdgcn_wmma_f32_16x16x32_bf16(
                false, at, false, bt, (short)0, acc_r[mt], false, false);
            v16bf al = loadA(smemB, mt * 16, k0);
            acc_r[mt] = __builtin_amdgcn_wmma_f32_16x16x32_bf16(
                false, al, false, bl, (short)0, acc_r[mt], false, false);
        }
    }

    // ---- Epilogue: hx/hy = tanh(acc + rec + b) via hardware V_TANH_F32 ----
    const float bn = bvec[depth * H_DIM + n0 + lane16];
    float* outx = out + cell;          // c = 0 (h_x)
    float* outy = out + cell + MAT;    // c = 1 (h_y)
#pragma unroll
    for (int mt = 0; mt < 4; ++mt) {
#pragma unroll
        for (int v = 0; v < 8; ++v) {
            const int m = mt * 16 + v + 8 * half;   // C/D layout: M = vgpr + 8*half
            const size_t o = (size_t)m * H_DIM + n0 + lane16;
            const float r = acc_r[mt][v] + bn;
            outx[o] = HTANH(acc_x[mt][v] + r);
            outy[o] = HTANH(acc_y[mt][v] + r);
        }
    }
}

extern "C" void kernel_launch(void* const* d_in, const int* in_sizes, int n_in,
                              void* d_out, int out_size, void* d_ws, size_t ws_size,
                              hipStream_t stream) {
    (void)in_sizes; (void)n_in; (void)out_size; (void)ws_size;
    const float* src = (const float*)d_in[0];
    const float* trg = (const float*)d_in[1];
    const float* W   = (const float*)d_in[2];
    const float* U   = (const float*)d_in[3];
    const float* b   = (const float*)d_in[4];
    float* out = (float*)d_out;
    unsigned short* wt = (unsigned short*)d_ws;   // 768 KB bf16 weights

    // 1) Convert + transpose weights to bf16 [d][mat][N][K].
    {
        const int total = DEPTH_N * 3 * H_DIM * H_DIM;
        prep_weights_kernel<<<(total + 255) / 256, 256, 0, stream>>>(W, U, wt);
    }

    // 2) Wavefront over anti-diagonals; cells on a diagonal are independent.
    for (int d = 0; d < DEPTH_N; ++d) {
        for (int step = 0; step <= SRC_N + TRG_N - 2; ++step) {
            const int i_min = step > (TRG_N - 1) ? step - (TRG_N - 1) : 0;
            const int i_max = step < (SRC_N - 1) ? step : (SRC_N - 1);
            const int count = i_max - i_min + 1;
            grid_cell_kernel<<<count, 512, 0, stream>>>(src, trg, b, wt, out,
                                                        d, step, i_min);
        }
    }
}
